// RNNBlock_33208687132706
// MI455X (gfx1250) — matrix operand order repaired and checked by reading, not compiled, not run on hardware
//
#include <hip/hip_runtime.h>

typedef _Float16 f16;
typedef __attribute__((ext_vector_type(16))) _Float16 v16h;
typedef __attribute__((ext_vector_type(8)))  float    v8f;

#define T_LEN 784
#define BSZ   64
#define DH    512
#define M_TOT (T_LEN * BSZ)   // 50176
#define NKSTEP 16             // 512 / 32
#define BD    (BSZ * DH)      // 32768
#define CLEN  16              // scan chunk length
#define NCHUNK (T_LEN / CLEN) // 49

union FragU { uint4 u[2]; v16h v; };

// ---------------- initial dense: h = relu(x @ W0 + b0), f16 out ----------------
__global__ __launch_bounds__(256) void init_dense_kernel(
    const float* __restrict__ x, const float* __restrict__ W0,
    const float* __restrict__ b0, f16* __restrict__ h)
{
    const int tid = blockIdx.x * 256 + threadIdx.x;   // m*512 + d
    const int d = tid & (DH - 1);
    const int m = tid >> 9;
    const float x0 = x[m * 2 + 0];
    const float x1 = x[m * 2 + 1];
    const float v = fmaf(x0, W0[d], fmaf(x1, W0[DH + d], b0[d]));
    h[(size_t)m * DH + d] = (f16)fmaxf(v, 0.f);
}

// ---------- convert f32 weights -> f16, transposed to [N][K] (for WMMA B frags) ----------
__global__ __launch_bounds__(256) void convert_wT_kernel(
    const float* __restrict__ Bw, const float* __restrict__ Cw,
    f16* __restrict__ BtH, f16* __restrict__ CtH)
{
    __shared__ float tile[32][33];
    const int mat = blockIdx.z;                 // 0..5 : 3 B mats then 3 C mats
    const float* src = (mat < 3) ? (Bw + (size_t)mat * DH * DH)
                                 : (Cw + (size_t)(mat - 3) * DH * DH);
    f16* dst = (mat < 3) ? (BtH + (size_t)mat * DH * DH)
                         : (CtH + (size_t)(mat - 3) * DH * DH);
    const int tx = threadIdx.x;                 // 0..31
    const int ty = threadIdx.y;                 // 0..7
    const int kbase = blockIdx.y * 32;
    const int nbase = blockIdx.x * 32;
#pragma unroll
    for (int j = 0; j < 32; j += 8)
        tile[ty + j][tx] = src[(size_t)(kbase + ty + j) * DH + nbase + tx];
    __syncthreads();
#pragma unroll
    for (int j = 0; j < 32; j += 8)
        dst[(size_t)(nbase + ty + j) * DH + kbase + tx] = (f16)tile[tx][ty + j];
}

// ---------------- main GEMM: [M,512](f16) x [512,512](f16, B given transposed [N][K]) ----------------
// 256 threads = 8 waves (wave32). Block tile 128M x 128N, K-step 32, double-buffered LDS.
// Each wave: 2 m-tiles x 4 n-tiles of 16x16 (v_wmma_f32_16x16x32_f16).
template <bool RELU_OUT_F16>
__global__ __launch_bounds__(256) void gemm512_wmma(
    const f16* __restrict__ A, const f16* __restrict__ Bt,
    float* __restrict__ outF, f16* __restrict__ outH)
{
    __shared__ f16 Als[2][128][40];   // padded stride 40 elems (80B) to spread LDS banks
    __shared__ f16 Bls[2][128][40];

    const int tid  = threadIdx.x;
    const int lane = tid & 31;
    const int wave = tid >> 5;
    const int waveM = wave & 3;       // 0..3
    const int waveN = wave >> 2;      // 0..1
    const int m0 = blockIdx.y * 128;
    const int n0 = blockIdx.x * 128;

    // global->LDS staging: each thread moves 16 f16 of A and 16 f16 of Bt per k-step
    const int lr = tid >> 1;          // tile row 0..127
    const int lc = (tid & 1) << 4;    // col segment 0 or 16

    const f16* Ag = A  + (size_t)(m0 + lr) * DH + lc;
    const f16* Bg = Bt + (size_t)(n0 + lr) * DH + lc;

    v8f acc[2][4];
#pragma unroll
    for (int mt = 0; mt < 2; ++mt)
#pragma unroll
        for (int nt = 0; nt < 4; ++nt)
#pragma unroll
            for (int i = 0; i < 8; ++i)
                acc[mt][nt][i] = 0.f;

    uint4 sA0 = *(const uint4*)Ag;
    uint4 sA1 = *(const uint4*)(Ag + 8);
    uint4 sB0 = *(const uint4*)Bg;
    uint4 sB1 = *(const uint4*)(Bg + 8);
    *(uint4*)&Als[0][lr][lc]     = sA0;
    *(uint4*)&Als[0][lr][lc + 8] = sA1;
    *(uint4*)&Bls[0][lr][lc]     = sB0;
    *(uint4*)&Bls[0][lr][lc + 8] = sB1;
    __syncthreads();

    // WMMA fragment LDS addressing (ISA 16-bit A 16x32 / B 32x16 layouts):
    // A: lanes 0-15 hold row (lane), K0-7 + K16-23 ; lanes 16-31 hold K8-15 + K24-31
    // B: lanes 0-15 hold col (lane), K0-7 + K16-23 ; lanes 16-31 hold K8-15 + K24-31
    const int half = lane >> 4;
    const int l15  = lane & 15;
    const int aRow0 = waveM * 32 + l15;
    const int bRow0 = waveN * 64 + l15;
    const int aOff  = half * 8;       // elems
    const int bOff  = half * 8;       // elems

#pragma unroll 1
    for (int kk = 0; kk < NKSTEP; ++kk) {
        const int cur = kk & 1;
        if (kk + 1 < NKSTEP) {
            const f16* Ag2 = Ag + (kk + 1) * 32;
            const f16* Bg2 = Bg + (kk + 1) * 32;
            sA0 = *(const uint4*)Ag2;
            sA1 = *(const uint4*)(Ag2 + 8);
            sB0 = *(const uint4*)Bg2;
            sB1 = *(const uint4*)(Bg2 + 8);
        }

        FragU afr[2], bfr[4];
#pragma unroll
        for (int mt = 0; mt < 2; ++mt) {
            const f16* p = &Als[cur][aRow0 + mt * 16][aOff];
            afr[mt].u[0] = *(const uint4*)p;          // K pairs 0-7 (or 8-15)
            afr[mt].u[1] = *(const uint4*)(p + 16);   // K pairs 16-23 (or 24-31)
        }
#pragma unroll
        for (int nt = 0; nt < 4; ++nt) {
            const f16* p = &Bls[cur][bRow0 + nt * 16][bOff];
            bfr[nt].u[0] = *(const uint4*)p;          // K 0-7  (or 8-15)
            bfr[nt].u[1] = *(const uint4*)(p + 16);   // K 16-23 (or 24-31)
        }

#pragma unroll
        for (int mt = 0; mt < 2; ++mt)
#pragma unroll
            for (int nt = 0; nt < 4; ++nt)
                acc[mt][nt] = __builtin_amdgcn_wmma_f32_16x16x32_f16(
                    false, afr[mt].v, false, bfr[nt].v,
                    (short)0, acc[mt][nt], false, false);

        if (kk + 1 < NKSTEP) {
            const int nxt = cur ^ 1;
            *(uint4*)&Als[nxt][lr][lc]     = sA0;
            *(uint4*)&Als[nxt][lr][lc + 8] = sA1;
            *(uint4*)&Bls[nxt][lr][lc]     = sB0;
            *(uint4*)&Bls[nxt][lr][lc + 8] = sB1;
            __syncthreads();
        }
    }

    // Epilogue. D layout: VGPR i -> row i (lanes 0-15) or row 8+i (lanes 16-31), col = lane&15.
#pragma unroll
    for (int mt = 0; mt < 2; ++mt)
#pragma unroll
        for (int nt = 0; nt < 4; ++nt)
#pragma unroll
            for (int i = 0; i < 8; ++i) {
                const int row = m0 + waveM * 32 + mt * 16 + half * 8 + i;
                const int col = n0 + waveN * 64 + nt * 16 + l15;
                const size_t idx = (size_t)row * DH + col;
                const float v = acc[mt][nt][i];
                if (RELU_OUT_F16) outH[idx] = (f16)fmaxf(v, 0.f);
                else              outF[idx] = v;
            }
}

// ---------------- chunked parallel scan: h_t = a*h_{t-1} + u_t ----------------
// Phase 1: per (chunk,b,d) compute chunk carry with h_in = 0 (16 independent loads in flight).
__global__ __launch_bounds__(256) void scan_phase1_kernel(
    const float* __restrict__ u, const float* __restrict__ Adiag,
    float* __restrict__ carry, int layer)
{
    const int tid = blockIdx.x * 256 + threadIdx.x;   // c*BD + b*512 + d
    const int bd = tid & (BD - 1);
    const int c  = tid >> 15;                         // BD == 1<<15
    const float a = Adiag[layer * DH + (bd & (DH - 1))];
    const float* up = u + (size_t)c * CLEN * BD + bd;
    float h = 0.f;
#pragma unroll
    for (int i = 0; i < CLEN; ++i)
        h = fmaf(h, a, up[(size_t)i * BD]);
    carry[tid] = h;
}

// Phase 2: exclusive scan of the 49 chunk carries per (b,d), multiplier a^CLEN.
__global__ __launch_bounds__(256) void scan_phase2_kernel(
    float* __restrict__ carry, const float* __restrict__ Adiag, int layer)
{
    const int bd = blockIdx.x * 256 + threadIdx.x;    // 32768 threads
    const float a = Adiag[layer * DH + (bd & (DH - 1))];
    float ap = a * a;                                 // a^2
    ap = ap * ap;                                     // a^4
    ap = ap * ap;                                     // a^8
    ap = ap * ap;                                     // a^16 == a^CLEN
    float h = 0.f;
#pragma unroll 1
    for (int c = 0; c < NCHUNK; ++c) {
        const size_t idx = (size_t)c * BD + bd;
        const float t = carry[idx];
        carry[idx] = h;                               // exclusive prefix = h_in for chunk c
        h = fmaf(h, ap, t);
    }
}

// Phase 3: replay each chunk from its exact carry-in, emit f16 hs.
__global__ __launch_bounds__(256) void scan_phase3_kernel(
    const float* __restrict__ u, const float* __restrict__ Adiag,
    const float* __restrict__ carry, f16* __restrict__ hs, int layer)
{
    const int tid = blockIdx.x * 256 + threadIdx.x;
    const int bd = tid & (BD - 1);
    const int c  = tid >> 15;
    const float a = Adiag[layer * DH + (bd & (DH - 1))];
    const size_t base = (size_t)c * CLEN * BD + bd;
    float h = carry[tid];
#pragma unroll
    for (int i = 0; i < CLEN; ++i) {
        h = fmaf(h, a, u[base + (size_t)i * BD]);
        hs[base + (size_t)i * BD] = (f16)h;
    }
}

// ---------------- final dense: out = h @ Wf + bf (one wave32 per row) ----------------
__global__ __launch_bounds__(256) void final_dense_kernel(
    const f16* __restrict__ h, const float* __restrict__ Wf,
    const float* __restrict__ bf, float* __restrict__ out)
{
    const int lane = threadIdx.x & 31;
    const int m = blockIdx.x * 8 + (threadIdx.x >> 5);
    float s0 = 0.f, s1 = 0.f;
#pragma unroll
    for (int j = 0; j < 16; ++j) {
        const int dd = lane + j * 32;
        const float hv = (float)h[(size_t)m * DH + dd];
        s0 = fmaf(hv, Wf[dd * 2 + 0], s0);
        s1 = fmaf(hv, Wf[dd * 2 + 1], s1);
    }
#pragma unroll
    for (int off = 16; off > 0; off >>= 1) {
        s0 += __shfl_down(s0, off, 32);
        s1 += __shfl_down(s1, off, 32);
    }
    if (lane == 0) {
        out[m * 2 + 0] = s0 + bf[0];
        out[m * 2 + 1] = s1 + bf[1];
    }
}

extern "C" void kernel_launch(void* const* d_in, const int* in_sizes, int n_in,
                              void* d_out, int out_size, void* d_ws, size_t ws_size,
                              hipStream_t stream)
{
    const float* x  = (const float*)d_in[0];
    const float* W0 = (const float*)d_in[1];
    const float* b0 = (const float*)d_in[2];
    const float* Ad = (const float*)d_in[3];
    const float* Bw = (const float*)d_in[4];
    const float* Cw = (const float*)d_in[5];
    const float* Wf = (const float*)d_in[6];
    const float* bf = (const float*)d_in[7];
    float* out = (float*)d_out;

    char* ws = (char*)d_ws;
    const size_t WMAT = (size_t)3 * DH * DH;                 // elems per weight set
    f16* BtH = (f16*)ws;                                     // [3][N][K] f16
    f16* CtH = BtH + WMAT;                                   // [3][N][K] f16
    size_t off = 2 * WMAT * sizeof(f16);
    f16*   R0 = (f16*)(ws + off);  off += (size_t)M_TOT * DH * sizeof(f16);   // acts f16
    f16*   R1 = (f16*)(ws + off);  off += (size_t)M_TOT * DH * sizeof(f16);   // acts f16
    float* R2 = (float*)(ws + off);  off += (size_t)M_TOT * DH * sizeof(float); // u f32
    float* CAR = (float*)(ws + off);                         // [NCHUNK][BD] carries
    (void)in_sizes; (void)n_in; (void)out_size; (void)ws_size;

    convert_wT_kernel<<<dim3(16, 16, 6), dim3(32, 8), 0, stream>>>(Bw, Cw, BtH, CtH);
    init_dense_kernel<<<(M_TOT * DH) / 256, 256, 0, stream>>>(x, W0, b0, R0);

    const dim3 gg(DH / 128, M_TOT / 128);   // (4, 392)
    const int scanBlocks = (NCHUNK * BD) / 256;   // 6272
    for (int l = 0; l < 3; ++l) {
        gemm512_wmma<false><<<gg, 256, 0, stream>>>(R0, BtH + (size_t)l * DH * DH, R2, nullptr);
        scan_phase1_kernel<<<scanBlocks, 256, 0, stream>>>(R2, Ad, CAR, l);
        scan_phase2_kernel<<<BD / 256, 256, 0, stream>>>(CAR, Ad, l);
        scan_phase3_kernel<<<scanBlocks, 256, 0, stream>>>(R2, Ad, CAR, R1, l);
        gemm512_wmma<true><<<gg, 256, 0, stream>>>(R1, CtH + (size_t)l * DH * DH, nullptr, R0);
    }

    final_dense_kernel<<<M_TOT / 8, 256, 0, stream>>>(R0, Wf, bf, out);
}
